// A2M_4604204941662
// MI455X (gfx1250) — compile-verified, matrix operand order checked by reading, not run on hardware
//
#include <hip/hip_runtime.h>

// Fused Actor->Map attention (LaneGCN A2M-style) for MI455X / gfx1250.
// One block = 16 lane nodes, 256 threads = 8 wave32 waves, each wave owns a
// 16-wide slice of C=128. All GEMMs run on v_wmma_f32_16x16x32_f16 with f32
// accumulation. GroupNorm is a fully parallel register+LDS reduction; the
// [B,M,A,C] pair intermediates never touch HBM.

#define DEVINL __device__ __forceinline__

typedef __attribute__((ext_vector_type(16))) _Float16 v16h;
typedef __attribute__((ext_vector_type(8)))  float    v8f;

constexpr int Bsz = 4, Msz = 1500, Asz = 32, Csz = 128;
constexpr int MT  = (Msz + 15) / 16;     // 94 m-tiles per scene
constexpr float DIST2 = 49.0f;           // 7.0^2

// K index inside a 32-wide K-tile for half-element h of a 16-half fragment,
// matching the CDNA5 16-bit A/B VGPR striping (CDNA5 ISA 7.12.2).
// Consecutive h within an octet are K-contiguous -> vectorizes to b128 loads.
DEVINL int kmap(int h, int lane) { return (h & 7) + 8 * (2 * (h >> 3) + (lane >> 4)); }

// A fragment: 16x32 f16 slice of a row-major [16 x 128] f16 LDS tile.
DEVINL v16h load_afrag(const _Float16* ab, int kt, int lane) {
  v16h a; int m = lane & 15; int kb = kt * 32;
  #pragma unroll
  for (int h = 0; h < 16; ++h) a[h] = ab[m * Csz + kb + kmap(h, lane)];
  return a;
}

// B fragment: 32x16 slice of W^T where W is row-major [128][ldk] f32 in global
// memory (weight[out][in]); B[k][n] = W[n][koff + k]. Converted to f16 on load.
DEVINL v16h load_bfrag(const float* W, int ldk, int koff, int n0, int kt, int lane) {
  v16h b; int n = n0 + (lane & 15); int kb = koff + kt * 32;
  #pragma unroll
  for (int h = 0; h < 16; ++h) b[h] = (_Float16)W[n * ldk + kb + kmap(h, lane)];
  return b;
}

// [16,128] = [16,128(f16 LDS)] x [128,128(global f32 W^T)] -> f32 LDS
DEVINL void gemm_gW(const _Float16* ab, const float* W, int ldk, int koff,
                    float* ob, int tid) {
  int lane = tid & 31, n0 = (tid >> 5) * 16;
  v8f acc = {};
  #pragma unroll
  for (int kt = 0; kt < 4; ++kt) {
    v16h a = load_afrag(ab, kt, lane);
    v16h b = load_bfrag(W, ldk, koff, n0, kt, lane);
    acc = __builtin_amdgcn_wmma_f32_16x16x32_f16(false, a, false, b,
                                                 (short)0, acc, false, false);
  }
  int col = n0 + (lane & 15), rb = (lane >> 4) * 8;
  #pragma unroll
  for (int r = 0; r < 8; ++r) ob[(rb + r) * Csz + col] = acc[r];
}

// Same but with the per-wave B fragments preloaded in registers (hot loop).
DEVINL void gemm_frag(const _Float16* ab, const v16h* bf, float* ob, int tid) {
  int lane = tid & 31, n0 = (tid >> 5) * 16;
  v8f acc = {};
  #pragma unroll
  for (int kt = 0; kt < 4; ++kt) {
    v16h a = load_afrag(ab, kt, lane);
    acc = __builtin_amdgcn_wmma_f32_16x16x32_f16(false, a, false, bf[kt],
                                                 (short)0, acc, false, false);
  }
  int col = n0 + (lane & 15), rb = (lane >> 4) * 8;
  #pragma unroll
  for (int r = 0; r < 8; ++r) ob[(rb + r) * Csz + col] = acc[r];
}

// Parallel GroupNorm(1 group) over C=128 for 16 rows. Each thread owns 8
// register values of row r = tid&15; 16 partials per row are folded in LDS.
// vals[] is normalized in place. NOTE: no trailing barrier -- every call site
// has a __syncthreads() (via the next GEMM) before red_s/red_q/stats are
// written again.
DEVINL void gn_apply(float vals[8], float* red_s, float* red_q, float* stats,
                     const float* g, const float* bt, bool relu, int tid) {
  const int r = tid & 15, gi = tid >> 4, gc = gi * 8;
  float s = 0.f, q = 0.f;
  #pragma unroll
  for (int j = 0; j < 8; ++j) { s += vals[j]; q += vals[j] * vals[j]; }
  red_s[r * 16 + gi] = s;
  red_q[r * 16 + gi] = q;
  __syncthreads();
  if (tid < 16) {
    float ss = 0.f, qq = 0.f;
    #pragma unroll
    for (int k = 0; k < 16; ++k) { ss += red_s[tid * 16 + k]; qq += red_q[tid * 16 + k]; }
    float mu  = ss * (1.0f / Csz);
    float var = qq * (1.0f / Csz) - mu * mu;
    stats[tid * 2]     = mu;
    stats[tid * 2 + 1] = rsqrtf(fmaxf(var, 0.f) + 1e-5f);
  }
  __syncthreads();
  float mu = stats[r * 2], rs = stats[r * 2 + 1];
  #pragma unroll
  for (int j = 0; j < 8; ++j) {
    float v = (vals[j] - mu) * rs * g[gc + j] + bt[gc + j];
    vals[j] = relu ? fmaxf(v, 0.f) : v;
  }
}

// Warm L2/WGP$ with a weight matrix (one cacheline per prefetch per thread).
DEVINL void prefetch_bytes(const float* p, int bytes, int tid) {
  const char* c = (const char*)p;
  for (int off = tid * 128; off < bytes; off += 256 * 128)
    __builtin_prefetch(c + off, 0, 1);
}

struct P {
  const float *feat, *turn, *control, *intersect, *ctrs, *actors, *actor_ctrs;
  const float *meta_w, *meta_g, *meta_b;
  const float *dist_w1, *dist_b1, *dist_w2, *dist_g, *dist_bt;
  const float *q_w, *q_g, *q_b;
  const float *ctx_w1, *ctx_g, *ctx_b, *ctx_w2;
  const float *agt_w, *norm_g, *norm_b;
  const float *lin_w, *lin_g, *lin_b;
  float* out;
};

__global__ __launch_bounds__(256) void fused_a2m(P p) {
  __shared__ _Float16 abuf[16 * Csz];     // f16 WMMA A staging        (4 KB)
  __shared__ float    cbuf[16 * Csz];     // f32 GEMM output staging   (8 KB)
  __shared__ _Float16 acbuf[Asz * Csz];   // actors @ Wc^T, f16        (8 KB)
  __shared__ float    red_s[16 * 16];     // GN partial sums           (1 KB)
  __shared__ float    red_q[16 * 16];     // GN partial sumsq          (1 KB)
  __shared__ float    stats[32];          // GN per-row mean/rstd
  __shared__ float    sctr[16 * 2];       // lane-node centers
  __shared__ float    sact[Asz * 2];      // actor centers

  const int tid  = threadIdx.x;
  const int b    = blockIdx.y, mt = blockIdx.x;
  const int r    = tid & 15;              // owned row   (0..15)
  const int gc   = (tid >> 4) * 8;        // owned col*8 (0..120)
  const int lane = tid & 31;

  const int  mrow  = mt * 16 + r;
  const int  rl    = min(mrow, Msz - 1);  // clamp edge rows; store is masked
  const bool valid = (mrow < Msz);

  if (tid < 16) {
    int mr = min(mt * 16 + tid, Msz - 1);
    sctr[tid * 2 + 0] = p.ctrs[(b * Msz + mr) * 2 + 0];
    sctr[tid * 2 + 1] = p.ctrs[(b * Msz + mr) * 2 + 1];
  }
  if (tid < Asz) {
    sact[tid * 2 + 0] = p.actor_ctrs[(b * Asz + tid) * 2 + 0];
    sact[tid * 2 + 1] = p.actor_ctrs[(b * Asz + tid) * 2 + 1];
  }

  float x_reg[8], res_reg[8], qc_reg[8], msg_reg[8], vals[8];

  // ---------------- stage 0: meta fuse -----------------------------------
  #pragma unroll
  for (int j = 0; j < 8; ++j)
    abuf[r * Csz + gc + j] = (_Float16)p.feat[(b * Msz + rl) * Csz + gc + j];
  __syncthreads();
  gemm_gW(abuf, p.meta_w, Csz + 4, 0, cbuf, tid);     // K=128 part via WMMA
  __syncthreads();
  {                                                   // K=4 meta tail via VALU
    float t0 = p.turn[(b * Msz + rl) * 2 + 0], t1 = p.turn[(b * Msz + rl) * 2 + 1];
    float ct = p.control[b * Msz + rl], it = p.intersect[b * Msz + rl];
    #pragma unroll
    for (int j = 0; j < 8; ++j) {
      int n = gc + j; const float* w = p.meta_w + n * (Csz + 4) + Csz;
      vals[j] = cbuf[r * Csz + n] +
                t0 * w[0] + t1 * w[1] + ct * w[2] + it * w[3];
    }
  }
  gn_apply(vals, red_s, red_q, stats, p.meta_g, p.meta_b, true, tid);
  #pragma unroll
  for (int j = 0; j < 8; ++j) x_reg[j] = vals[j];

  // ---------------- two Att(n_map, n_actor) blocks ------------------------
  for (int i = 0; i < 2; ++i) {
    const float* dw1 = p.dist_w1 + i * Csz * 2;  const float* db1 = p.dist_b1 + i * Csz;
    const float* dw2 = p.dist_w2 + i * Csz * Csz;
    const float* dg  = p.dist_g  + i * Csz;      const float* dbt = p.dist_bt + i * Csz;
    const float* qw  = p.q_w     + i * Csz * Csz;
    const float* qg  = p.q_g     + i * Csz;      const float* qb  = p.q_b     + i * Csz;
    const float* cw1 = p.ctx_w1  + i * Csz * 3 * Csz;
    const float* cg  = p.ctx_g   + i * Csz;      const float* cb_ = p.ctx_b   + i * Csz;
    const float* cw2 = p.ctx_w2  + i * Csz * Csz;
    const float* aw  = p.agt_w   + i * Csz * Csz;
    const float* ng  = p.norm_g  + i * Csz;      const float* nb  = p.norm_b  + i * Csz;
    const float* lw  = p.lin_w   + i * Csz * Csz;
    const float* lg  = p.lin_g   + i * Csz;      const float* lb  = p.lin_b   + i * Csz;

    // Warm the cache for the hot-loop weights while the q/actor GEMMs run.
    prefetch_bytes(dw2, Csz * Csz * 4, tid);          // global_prefetch_b8
    prefetch_bytes(cw1, 3 * Csz * Csz * 4, tid);

    #pragma unroll
    for (int j = 0; j < 8; ++j) res_reg[j] = x_reg[j];

    // q = relu(GN(x @ qw^T)), then qc = q @ Wq^T (ctx_w1 cols C..2C)
    __syncthreads();
    #pragma unroll
    for (int j = 0; j < 8; ++j) abuf[r * Csz + gc + j] = (_Float16)x_reg[j];
    __syncthreads();
    gemm_gW(abuf, qw, Csz, 0, cbuf, tid);
    __syncthreads();
    #pragma unroll
    for (int j = 0; j < 8; ++j) vals[j] = cbuf[r * Csz + gc + j];
    gn_apply(vals, red_s, red_q, stats, qg, qb, true, tid);
    #pragma unroll
    for (int j = 0; j < 8; ++j) abuf[r * Csz + gc + j] = (_Float16)vals[j];
    __syncthreads();
    gemm_gW(abuf, cw1, 3 * Csz, Csz, cbuf, tid);
    __syncthreads();
    #pragma unroll
    for (int j = 0; j < 8; ++j) qc_reg[j] = cbuf[r * Csz + gc + j];

    // ac = actors @ Wc^T (ctx_w1 cols 2C..3C), stored f16, 2 half-tiles of 16
    for (int half = 0; half < 2; ++half) {
      __syncthreads();
      #pragma unroll
      for (int j = 0; j < 8; ++j)
        abuf[r * Csz + gc + j] =
            (_Float16)p.actors[(b * Asz + half * 16 + r) * Csz + gc + j];
      __syncthreads();
      gemm_gW(abuf, cw1, 3 * Csz, 2 * Csz, cbuf, tid);
      __syncthreads();
      #pragma unroll
      for (int j = 0; j < 8; ++j)
        acbuf[(half * 16 + r) * Csz + gc + j] = (_Float16)cbuf[r * Csz + gc + j];
    }
    __syncthreads();

    // Preload the two hot-loop weight matrices as per-wave WMMA B fragments:
    // dist_w2 and ctx_w1[:, :C] stay register-resident for all 32 actors.
    v16h w2f[4], wdf[4];
    {
      int n0 = (tid >> 5) * 16;
      #pragma unroll
      for (int kt = 0; kt < 4; ++kt) {
        w2f[kt] = load_bfrag(dw2, Csz,     0, n0, kt, lane);
        wdf[kt] = load_bfrag(cw1, 3 * Csz, 0, n0, kt, lane);
      }
    }

    // Hoist the dist-MLP first layer coefficients for this thread's 8 cols.
    float w1a[8], w1b[8], b1r[8];
    #pragma unroll
    for (int j = 0; j < 8; ++j) {
      int n = gc + j;
      w1a[j] = dw1[n * 2 + 0];
      w1b[j] = dw1[n * 2 + 1];
      b1r[j] = db1[n];
    }

    #pragma unroll
    for (int j = 0; j < 8; ++j) msg_reg[j] = 0.f;

    // -------- per-actor pair pipeline (the 25 GFLOP hot loop) -------------
    for (int a = 0; a < Asz; ++a) {
      float d0 = sctr[r * 2]     - sact[a * 2];
      float d1 = sctr[r * 2 + 1] - sact[a * 2 + 1];
      // h = relu(d @ W1^T + b1)  (K=2, VALU)
      #pragma unroll
      for (int j = 0; j < 8; ++j)
        abuf[r * Csz + gc + j] =
            (_Float16)fmaxf(d0 * w1a[j] + d1 * w1b[j] + b1r[j], 0.f);
      __syncthreads();
      gemm_frag(abuf, w2f, cbuf, tid);                  // h @ W2^T
      __syncthreads();
      #pragma unroll
      for (int j = 0; j < 8; ++j) vals[j] = cbuf[r * Csz + gc + j];
      gn_apply(vals, red_s, red_q, stats, dg, dbt, true, tid);  // distf
      #pragma unroll
      for (int j = 0; j < 8; ++j) abuf[r * Csz + gc + j] = (_Float16)vals[j];
      __syncthreads();
      gemm_frag(abuf, wdf, cbuf, tid);                  // distf @ Wd^T
      __syncthreads();
      #pragma unroll
      for (int j = 0; j < 8; ++j)                        // + q-term + actor-term
        vals[j] = cbuf[r * Csz + gc + j] + qc_reg[j] +
                  (float)acbuf[a * Csz + gc + j];
      gn_apply(vals, red_s, red_q, stats, cg, cb_, true, tid);  // h2
      float msk = (d0 * d0 + d1 * d1 <= DIST2) ? 1.f : 0.f;
      #pragma unroll
      for (int j = 0; j < 8; ++j) msg_reg[j] += msk * vals[j];
    }

    // msg2 = msg @ cw2^T
    __syncthreads();
    #pragma unroll
    for (int j = 0; j < 8; ++j) abuf[r * Csz + gc + j] = (_Float16)msg_reg[j];
    __syncthreads();
    gemm_gW(abuf, cw2, Csz, 0, cbuf, tid);
    __syncthreads();
    #pragma unroll
    for (int j = 0; j < 8; ++j) msg_reg[j] = cbuf[r * Csz + gc + j];

    // x = relu(GN(x @ aw^T + msg2))
    #pragma unroll
    for (int j = 0; j < 8; ++j) abuf[r * Csz + gc + j] = (_Float16)x_reg[j];
    __syncthreads();
    gemm_gW(abuf, aw, Csz, 0, cbuf, tid);
    __syncthreads();
    #pragma unroll
    for (int j = 0; j < 8; ++j) vals[j] = cbuf[r * Csz + gc + j] + msg_reg[j];
    gn_apply(vals, red_s, red_q, stats, ng, nb, true, tid);
    #pragma unroll
    for (int j = 0; j < 8; ++j) x_reg[j] = vals[j];

    // x = relu(GN(x @ lw^T) + res)
    #pragma unroll
    for (int j = 0; j < 8; ++j) abuf[r * Csz + gc + j] = (_Float16)x_reg[j];
    __syncthreads();
    gemm_gW(abuf, lw, Csz, 0, cbuf, tid);
    __syncthreads();
    #pragma unroll
    for (int j = 0; j < 8; ++j) vals[j] = cbuf[r * Csz + gc + j];
    gn_apply(vals, red_s, red_q, stats, lg, lb, false, tid);
    #pragma unroll
    for (int j = 0; j < 8; ++j) x_reg[j] = fmaxf(vals[j] + res_reg[j], 0.f);
    __syncthreads();
  }

  if (valid) {
    #pragma unroll
    for (int j = 0; j < 8; ++j)
      p.out[(b * Msz + mrow) * Csz + gc + j] = x_reg[j];
  }
}

extern "C" void kernel_launch(void* const* d_in, const int* /*in_sizes*/, int /*n_in*/,
                              void* d_out, int /*out_size*/, void* /*d_ws*/,
                              size_t /*ws_size*/, hipStream_t stream) {
  P p;
  p.feat       = (const float*)d_in[0];
  p.turn       = (const float*)d_in[1];
  p.control    = (const float*)d_in[2];
  p.intersect  = (const float*)d_in[3];
  p.ctrs       = (const float*)d_in[4];
  p.actors     = (const float*)d_in[5];
  p.actor_ctrs = (const float*)d_in[6];
  p.meta_w     = (const float*)d_in[7];
  p.meta_g     = (const float*)d_in[8];
  p.meta_b     = (const float*)d_in[9];
  p.dist_w1    = (const float*)d_in[10];
  p.dist_b1    = (const float*)d_in[11];
  p.dist_w2    = (const float*)d_in[12];
  p.dist_g     = (const float*)d_in[13];
  p.dist_bt    = (const float*)d_in[14];
  p.q_w        = (const float*)d_in[15];
  p.q_g        = (const float*)d_in[16];
  p.q_b        = (const float*)d_in[17];
  p.ctx_w1     = (const float*)d_in[18];
  p.ctx_g      = (const float*)d_in[19];
  p.ctx_b      = (const float*)d_in[20];
  p.ctx_w2     = (const float*)d_in[21];
  p.agt_w      = (const float*)d_in[22];
  p.norm_g     = (const float*)d_in[23];
  p.norm_b     = (const float*)d_in[24];
  p.lin_w      = (const float*)d_in[25];
  p.lin_g      = (const float*)d_in[26];
  p.lin_b      = (const float*)d_in[27];
  p.out        = (float*)d_out;

  dim3 grid(MT, Bsz);
  hipLaunchKernelGGL(fused_a2m, grid, dim3(256), 0, stream, p);
}
